// AttnBlock_65060164600030
// MI455X (gfx1250) — compile-verified, matrix operand order checked by reading, not run on hardware
//
#include <hip/hip_runtime.h>
#include <hip/hip_bf16.h>

typedef __attribute__((ext_vector_type(16))) __bf16 v16bf;
typedef __attribute__((ext_vector_type(8)))  float  v8f;

#define DIN   2048
#define LCT   512
#define NQR   16
#define DOUT  1024
#define NBATCH 128
#define CIN   1024
#define HWSP  256   // 16*16

// D = A*B + C, bf16 inputs, fp32 accum (v_wmma_f32_16x16x32_bf16)
static __device__ __forceinline__ v8f wmma_bf16(v16bf a, v16bf b, v8f c) {
  return __builtin_amdgcn_wmma_f32_16x16x32_bf16(false, a, false, b, (short)0, c,
                                                 false, false);
}

// 16-bit A/B operand layout (ISA 7.12.2): lane (0-15)=row/col M, lane hi half
// shifts K by 8; element j holds K = k0 + (j>=8?16:0) + hi*8 + (j&7).
// K is contiguous in memory -> two 16B chunks per lane.
static __device__ __forceinline__ v16bf frag_from_bf16(const __bf16* base, int stride,
                                                       int row, int k0, int hi) {
  const __bf16* p = base + (size_t)row * stride + k0 + hi * 8;
  v16bf r;
#pragma unroll
  for (int j = 0; j < 8; ++j) r[j] = p[j];
#pragma unroll
  for (int j = 0; j < 8; ++j) r[8 + j] = p[16 + j];
  return r;
}

static __device__ __forceinline__ v16bf frag_from_f32(const float* base, int stride,
                                                      int row, int k0, int hi) {
  const float* p = base + (size_t)row * stride + k0 + hi * 8;
  v16bf r;
#pragma unroll
  for (int j = 0; j < 8; ++j) r[j] = (__bf16)p[j];
#pragma unroll
  for (int j = 0; j < 8; ++j) r[8 + j] = (__bf16)p[16 + j];
  return r;
}

// ---------------- K1: q[16,2048] = query @ W_in^T  (batch-invariant) --------
__global__ void k_qproj(const float* __restrict__ query, const float* __restrict__ W_in,
                        __bf16* __restrict__ qbf) {
  const int lane = threadIdx.x & 31;
  const int col = lane & 15, hi = lane >> 4;
  const int e0 = blockIdx.x * 16;
  v8f acc = {};
  for (int k0 = 0; k0 < DIN; k0 += 32) {
    // keep the next chunk of this lane's W_in row warm (global_prefetch_b8)
    if (k0 + 128 < DIN)
      __builtin_prefetch(W_in + (size_t)(e0 + col) * DIN + k0 + 128, 0, 3);
    v16bf a = frag_from_f32(query, DIN, col, k0, hi);        // A[qi, d]
    v16bf b = frag_from_f32(W_in, DIN, e0 + col, k0, hi);    // B[d, e] = W_in[e, d]
    acc = wmma_bf16(a, b, acc);
  }
#pragma unroll
  for (int v = 0; v < 8; ++v) {
    int m = v + hi * 8;                                      // qi
    qbf[m * DIN + e0 + col] = (__bf16)acc[v];
  }
}

// ---------------- K2: fused scores -> softmax -> mix per batch n ------------
__global__ void __launch_bounds__(256) k_attn(const float* __restrict__ context,
                                              const __bf16* __restrict__ qbf,
                                              __bf16* __restrict__ mixb) {
  __shared__ __bf16 ctx_s[LCT * 32];   // [l][d_local] bf16 tile, 32 KB
  __shared__ float  sc_s[NQR * LCT];   // scores fp32, 32 KB
  __bf16* w_s = ctx_s;                 // weights [16][512] bf16 reuse ctx_s after pass1

  const int tid = threadIdx.x, wave = tid >> 5, lane = tid & 31;
  const int col = lane & 15, hi = lane >> 4;
  const int n = blockIdx.x;
  const float* ctxn = context + (size_t)n * DIN * LCT;       // [d][l]

  // pass 1: scores[q,l] = sum_d q[q,d] * ctx[n,d,l]
  v8f acc[4];
#pragma unroll
  for (int t = 0; t < 4; ++t) acc[t] = (v8f){};
  for (int d0 = 0; d0 < DIN; d0 += 32) {
    __syncthreads();
    // stage 32 d-rows as bf16 [l][d_local]; b128 global loads, coalesced
    const float4* ctx4 = (const float4*)(ctxn + (size_t)d0 * LCT);
    for (int idx = tid; idx < 32 * (LCT / 4); idx += 256) {
      int dl = idx >> 7;        // row within tile
      int c4 = idx & 127;       // float4 index within row
      float4 f = ctx4[dl * 128 + c4];
      int l = c4 * 4;
      ctx_s[(l + 0) * 32 + dl] = (__bf16)f.x;
      ctx_s[(l + 1) * 32 + dl] = (__bf16)f.y;
      ctx_s[(l + 2) * 32 + dl] = (__bf16)f.z;
      ctx_s[(l + 3) * 32 + dl] = (__bf16)f.w;
    }
    // prefetch next 64KB tile while this one is consumed by the WMMAs
    if (d0 + 32 < DIN) {
      const float* nxt = ctxn + (size_t)(d0 + 32) * LCT;
#pragma unroll
      for (int t = 0; t < 2; ++t)
        __builtin_prefetch(nxt + (size_t)(tid + t * 256) * 32, 0, 3);
    }
    __syncthreads();
    v16bf a = frag_from_bf16(qbf, DIN, col, d0, hi);
#pragma unroll
    for (int t = 0; t < 4; ++t) {                            // each wave: 4 l-tiles
      int l0 = (wave * 4 + t) * 16;
      v16bf b = frag_from_bf16(ctx_s, 32, l0 + col, 0, hi);  // B[k,l] = ctx_s[l][k]
      acc[t] = wmma_bf16(a, b, acc[t]);
    }
  }
#pragma unroll
  for (int t = 0; t < 4; ++t)
#pragma unroll
    for (int v = 0; v < 8; ++v)
      sc_s[(v + hi * 8) * LCT + (wave * 4 + t) * 16 + col] = acc[t][v];
  __syncthreads();

  // softmax over l (512) — 2 rows per wave, wave32 shfl reductions
  for (int rr = 0; rr < 2; ++rr) {
    int r = wave * 2 + rr;
    float m = -3.4e38f;
    for (int l = lane; l < LCT; l += 32) m = fmaxf(m, sc_s[r * LCT + l]);
#pragma unroll
    for (int off = 16; off; off >>= 1) m = fmaxf(m, __shfl_xor(m, off, 32));
    float s = 0.f;
    for (int l = lane; l < LCT; l += 32) {
      float e = __expf(sc_s[r * LCT + l] - m);
      sc_s[r * LCT + l] = e;
      s += e;
    }
#pragma unroll
    for (int off = 16; off; off >>= 1) s += __shfl_xor(s, off, 32);
    float inv = 1.f / s;
    for (int l = lane; l < LCT; l += 32)
      w_s[r * LCT + l] = (__bf16)(sc_s[r * LCT + l] * inv);
  }
  __syncthreads();

  // pass 2: mix[q,d] = sum_l w[q,l] * ctx[n,d,l]  (context re-read hits L2)
  for (int t = 0; t < 16; ++t) {
    int d0 = (wave * 16 + t) * 16;
    v8f macc = {};
    for (int l0 = 0; l0 < LCT; l0 += 32) {
      v16bf a = frag_from_bf16(w_s, LCT, col, l0, hi);       // A[q, l]
      v16bf b = frag_from_f32(ctxn, LCT, d0 + col, l0, hi);  // B[l, d] = ctx[n,d,l]
      macc = wmma_bf16(a, b, macc);
    }
#pragma unroll
    for (int v = 0; v < 8; ++v)
      mixb[((size_t)n * NQR + (v + hi * 8)) * DIN + d0 + col] = (__bf16)macc[v];
  }
}

// ---------------- K3: out = tanh([mix|q] @ W_out^T), upsample-scatter -------
__global__ void k_out(const __bf16* __restrict__ mixb, const __bf16* __restrict__ qbf,
                      const float* __restrict__ W_out, float* __restrict__ out) {
  const int lane = threadIdx.x & 31;
  const int col = lane & 15, hi = lane >> 4;
  const int o0 = blockIdx.x * 16;
  const int n = blockIdx.y;
  const __bf16* An = mixb + (size_t)n * NQR * DIN;
  const float* Wrow = W_out + (size_t)(o0 + col) * (2 * DIN);
  v8f acc = {};
  for (int k0 = 0; k0 < DIN; k0 += 32) {                     // mix half
    if (k0 + 128 < DIN) __builtin_prefetch(Wrow + k0 + 128, 0, 3);
    v16bf a = frag_from_bf16(An, DIN, col, k0, hi);
    v16bf b = frag_from_f32(W_out, 2 * DIN, o0 + col, k0, hi);
    acc = wmma_bf16(a, b, acc);
  }
  for (int k0 = 0; k0 < DIN; k0 += 32) {                     // q half
    if (k0 + 128 < DIN) __builtin_prefetch(Wrow + DIN + k0 + 128, 0, 3);
    v16bf a = frag_from_bf16(qbf, DIN, col, k0, hi);
    v16bf b = frag_from_f32(W_out, 2 * DIN, o0 + col, DIN + k0, hi);
    acc = wmma_bf16(a, b, acc);
  }
  const int o = o0 + col;
#pragma unroll
  for (int v = 0; v < 8; ++v) {
    int qi = v + hi * 8;                                     // qi = hq*4 + wq
    float val = tanhf(acc[v]);
    float4 q4 = make_float4(val, val, val, val);
    float* p = out + ((size_t)n * (CIN + DOUT) + CIN + o) * HWSP
                   + (qi >> 2) * 64 + (qi & 3) * 4;          // nearest 4x4 block
#pragma unroll
    for (int h = 0; h < 4; ++h) *(float4*)(p + h * 16) = q4;
  }
}

// ---------------- K4: copy cat_tensor into channels [0,1024) ----------------
__global__ void k_copy(const float4* __restrict__ src, float4* __restrict__ dst) {
  const size_t total = (size_t)NBATCH * CIN * HWSP / 4;      // 8388608 float4
  for (size_t i = (size_t)blockIdx.x * blockDim.x + threadIdx.x; i < total;
       i += (size_t)gridDim.x * blockDim.x) {
    size_t n = i >> 16;            // 1024*256/4 = 65536 float4 per n
    size_t r = i & 65535;
    dst[n * 131072 + r] = src[i];  // dst row pitch = 2048*256/4
  }
}

extern "C" void kernel_launch(void* const* d_in, const int* in_sizes, int n_in,
                              void* d_out, int out_size, void* d_ws, size_t ws_size,
                              hipStream_t stream) {
  const float* cat     = (const float*)d_in[0];
  const float* context = (const float*)d_in[1];
  const float* query   = (const float*)d_in[2];
  const float* W_in    = (const float*)d_in[3];
  const float* W_out   = (const float*)d_in[4];
  float* out = (float*)d_out;

  __bf16* qbf  = (__bf16*)d_ws;                  // 16*2048 bf16 = 64 KB
  __bf16* mixb = qbf + (size_t)NQR * DIN;        // 128*16*2048 bf16 = 8 MB

  k_copy <<<dim3(2048),          dim3(256), 0, stream>>>((const float4*)cat, (float4*)out);
  k_qproj<<<dim3(DIN / 16),      dim3(32),  0, stream>>>(query, W_in, qbf);
  k_attn <<<dim3(NBATCH),        dim3(256), 0, stream>>>(context, qbf, mixb);
  k_out  <<<dim3(DOUT / 16, NBATCH), dim3(32), 0, stream>>>(mixb, qbf, W_out, out);
}
